// SeedGenerator_bak_22127671509189
// MI455X (gfx1250) — matrix operand order repaired and checked by reading, not compiled, not run on hardware
//
#include <hip/hip_runtime.h>
#include <hip/hip_bf16.h>

typedef __attribute__((ext_vector_type(16))) _Float16 v16h;
typedef __attribute__((ext_vector_type(8)))  float    v8f;

#define EB   16
#define EN   512
#define EFEAT 1024
#define ECIN 128
#define EDIM 64
#define EK   20
#define ENUP 1024
#define EEPS 1e-5f

#define XPITCH 48   // LDS pitch in halves (32 K + 16 pad): keeps v16h reads 32B aligned

// ---------------------------------------------------------------------------
// Fast per-batch WMMA GEMM:  out[b][o,s] (+)= act( sum_c W[o,c]*X[b][c,s] + bias[o] )
// REQUIRES: O % 64 == 0, C % 32 == 0, S % 64 == 0 (true for all large GEMMs here).
// Block = 128 threads = 4 waves. Each wave owns a 16(M) x 64(N) strip: 4 WMMA
// tiles sharing one A fragment per K-step. X tile (32K x 64N) is staged in a
// double-buffered LDS tile (one barrier per K-step), pre-converted to f16 and
// K-contiguous so each B fragment is one aligned v16h LDS read. All four B
// fragments are preloaded (single DS clause), then the four WMMAs issue
// back-to-back into the matrix pipe. A/B k = 16*(lane/16)+s (contiguous); C/D per CDNA5 ISA 7.12.2.
// ---------------------------------------------------------------------------
__global__ __launch_bounds__(128)
void gemm_wmma_fast(const float* __restrict__ W, const float* __restrict__ X,
                    const float* __restrict__ bias, float* __restrict__ out,
                    int O, int C, int S, int relu, int accum)
{
    __shared__ _Float16 xh[2][64 * XPITCH];   // double-buffered [col][k], 2 x 6 KB

    const int tid  = threadIdx.x;
    const int lane = tid & 31;
    const int wv   = tid >> 5;
    const int orow0 = (blockIdx.y * 4 + wv) * 16;
    const int scol0 = blockIdx.x * 64;
    const int b     = blockIdx.z;

    const float* __restrict__ Xb   = X   + (size_t)b * C * S + scol0;
    float*       __restrict__ outb = out + (size_t)b * O * S;

    const int am   = orow0 + (lane & 15);   // A row (M)
    const int akb  = (lane >> 4) << 4;      // A k base: 0 or 16
    const int bcol = lane & 15;             // B column within 16-wide subtile
    const int bkb  = (lane >> 4) << 4;      // B k base: 0 or 16

    // cooperative X-tile load: 128 threads -> 32 K-rows x 4 float4-groups
    const int xrow  = tid >> 2;             // K row within tile (0..31)
    const int xcolb = (tid & 3) << 4;       // first of 16 columns this thread stages

    const float* __restrict__ wp = W + (size_t)am * C;

    v8f acc[4] = {{}, {}, {}, {}};

    int bufi = 0;
    for (int k0 = 0; k0 < C; k0 += 32, bufi ^= 1) {
        // ---- coalesced global load of this thread's 16 columns of row k0+xrow
        const float* xrp = Xb + (size_t)(k0 + xrow) * S + xcolb;
        float4 xv0 = *(const float4*)(xrp + 0);
        float4 xv1 = *(const float4*)(xrp + 4);
        float4 xv2 = *(const float4*)(xrp + 8);
        float4 xv3 = *(const float4*)(xrp + 12);
        if (k0 + 32 < C)
            __builtin_prefetch(xrp + (size_t)32 * S, 0, 1);

        // ---- transposed f16 store into current buffer: xh[bufi][col][k]
        {
            _Float16* dst = &xh[bufi][0];
            float e[16] = { xv0.x, xv0.y, xv0.z, xv0.w,  xv1.x, xv1.y, xv1.z, xv1.w,
                            xv2.x, xv2.y, xv2.z, xv2.w,  xv3.x, xv3.y, xv3.z, xv3.w };
#pragma unroll
            for (int i = 0; i < 16; ++i)
                dst[(size_t)(xcolb + i) * XPITCH + xrow] = (_Float16)e[i];
        }
        __syncthreads();   // tile visible; prior-buffer reads finished one barrier ago

        // ---- A fragment: one contiguous 16-float run per lane (matches B)
        v16h a;
        {
            const float4 wa0 = *(const float4*)(wp + k0 + akb);
            const float4 wa1 = *(const float4*)(wp + k0 + akb + 4);
            const float4 wa2 = *(const float4*)(wp + k0 + akb + 8);
            const float4 wa3 = *(const float4*)(wp + k0 + akb + 12);
            a[0]  = (_Float16)wa0.x; a[1]  = (_Float16)wa0.y;
            a[2]  = (_Float16)wa0.z; a[3]  = (_Float16)wa0.w;
            a[4]  = (_Float16)wa1.x; a[5]  = (_Float16)wa1.y;
            a[6]  = (_Float16)wa1.z; a[7]  = (_Float16)wa1.w;
            a[8]  = (_Float16)wa2.x; a[9]  = (_Float16)wa2.y;
            a[10] = (_Float16)wa2.z; a[11] = (_Float16)wa2.w;
            a[12] = (_Float16)wa3.x; a[13] = (_Float16)wa3.y;
            a[14] = (_Float16)wa3.z; a[15] = (_Float16)wa3.w;
        }

        // ---- preload all 4 B fragments (one DS clause), then 4 WMMAs back-to-back
        const _Float16* src = &xh[bufi][0];
        v16h bf0 = *(const v16h*)&src[(size_t)( 0 + bcol) * XPITCH + bkb];
        v16h bf1 = *(const v16h*)&src[(size_t)(16 + bcol) * XPITCH + bkb];
        v16h bf2 = *(const v16h*)&src[(size_t)(32 + bcol) * XPITCH + bkb];
        v16h bf3 = *(const v16h*)&src[(size_t)(48 + bcol) * XPITCH + bkb];
        acc[0] = __builtin_amdgcn_wmma_f32_16x16x32_f16(false, a, false, bf0, (short)0, acc[0], false, false);
        acc[1] = __builtin_amdgcn_wmma_f32_16x16x32_f16(false, a, false, bf1, (short)0, acc[1], false, false);
        acc[2] = __builtin_amdgcn_wmma_f32_16x16x32_f16(false, a, false, bf2, (short)0, acc[2], false, false);
        acc[3] = __builtin_amdgcn_wmma_f32_16x16x32_f16(false, a, false, bf3, (short)0, acc[3], false, false);
    }

    // ---- epilogue: C/D layout (M = vgpr + 8*lanehi, N = lane%16)
    const int mhi = (lane >> 4) << 3;
#pragma unroll
    for (int vg = 0; vg < 8; ++vg) {
        const int m = orow0 + mhi + vg;
        const float bv = bias[m];
        float* prow = outb + (size_t)m * S + scol0 + bcol;
#pragma unroll
        for (int j = 0; j < 4; ++j) {
            float r = acc[j][vg] + bv;
            if (relu) r = fmaxf(r, 0.0f);
            if (accum) prow[j * 16] += r; else prow[j * 16] = r;
        }
    }
}

// ---------------------------------------------------------------------------
// Plain kernel for the two tiny odd-shaped matmuls (C=3 or O=3): one thread
// per output element; these are bandwidth-trivial and don't fit WMMA tiles.
// ---------------------------------------------------------------------------
__global__ void gemm_small(const float* __restrict__ W, const float* __restrict__ X,
                           const float* __restrict__ bias, float* __restrict__ out,
                           int O, int C, int S, int relu, size_t total)
{
    size_t t = (size_t)blockIdx.x * blockDim.x + threadIdx.x;
    if (t >= total) return;
    int s = (int)(t % S); size_t r = t / S;
    int o = (int)(r % O); int b = (int)(r / O);
    const float* xb = X + (size_t)b * C * S + s;
    const float* wr = W + (size_t)o * C;
    float acc = bias[o];
    for (int c = 0; c < C; ++c) acc += wr[c] * xb[(size_t)c * S];
    if (relu) acc = fmaxf(acc, 0.0f);
    out[t] = acc;
}

// ---------------------------------------------------------------------------
// Fold eval-mode BatchNorm into the preceding conv: w' = w*s, b' = (b-m)*s+beta
// ---------------------------------------------------------------------------
__global__ void fold_bn_kernel(const float* __restrict__ w, const float* __restrict__ b,
                               const float* __restrict__ g, const float* __restrict__ beta,
                               const float* __restrict__ m, const float* __restrict__ v,
                               float* __restrict__ w2, float* __restrict__ b2, int O, int C)
{
    size_t t = (size_t)blockIdx.x * blockDim.x + threadIdx.x;
    if (t >= (size_t)O * C) return;
    int o = (int)(t / C);
    float s = g[o] * rsqrtf(v[o] + EEPS);
    w2[t] = w[t] * s;
    if ((t % C) == 0) b2[o] = (b[o] - m[o]) * s + beta[o];
}

// cat([pf,pf], axis=1): out (B,256,N)
__global__ void concat_dup_kernel(float* __restrict__ out, const float* __restrict__ pf, size_t total)
{
    size_t t = (size_t)blockIdx.x * blockDim.x + threadIdx.x;
    if (t >= total) return;
    int n = (int)(t % EN); size_t r = t / EN;
    int c = (int)(r % 256); int b = (int)(r / 256);
    out[t] = pf[((size_t)b * ECIN + (c & 127)) * EN + n];
}

// brute-force kNN: nearest-first top-K (includes self), ties -> lower index
__global__ void knn_topk_kernel(const float* __restrict__ xyz, int* __restrict__ idx)
{
    int t = blockIdx.x * blockDim.x + threadIdx.x;
    if (t >= EB * EN) return;
    int b = t / EN, n = t % EN;
    const float* px = xyz + (size_t)b * 3 * EN;
    float xn = px[n], yn = px[EN + n], zn = px[2 * EN + n];
    float sqn = xn * xn + yn * yn + zn * zn;
    float bd[EK]; int bi[EK];
#pragma unroll
    for (int i = 0; i < EK; ++i) { bd[i] = 3.4e38f; bi[i] = 0; }
    for (int j = 0; j < EN; ++j) {
        float xj = px[j], yj = px[EN + j], zj = px[2 * EN + j];
        float d = sqn + (xj * xj + yj * yj + zj * zj)
                - 2.0f * (xn * xj + yn * yj + zn * zj);
        if (d < bd[EK - 1]) {
            int p = EK - 1;
            while (p > 0 && d < bd[p - 1]) { bd[p] = bd[p - 1]; bi[p] = bi[p - 1]; --p; }
            bd[p] = d; bi[p] = j;
        }
    }
    int* o = idx + ((size_t)b * EN + n) * EK;
#pragma unroll
    for (int i = 0; i < EK; ++i) o[i] = bi[i];
}

// out[b,c,n,k] = ctr[b,c,n] - src[b,c,idx[b,n,k]]
__global__ void gather_sub_kernel(float* __restrict__ out, const float* __restrict__ ctr,
                                  const float* __restrict__ src, const int* __restrict__ idx,
                                  int C, size_t total)
{
    size_t t = (size_t)blockIdx.x * blockDim.x + threadIdx.x;
    if (t >= total) return;
    int k = (int)(t % EK); size_t r = t / EK;
    int n = (int)(r % EN); r /= EN;
    int c = (int)(r % C);  int b = (int)(r / C);
    int j = idx[((size_t)b * EN + n) * EK + k];
    out[t] = ctr[((size_t)b * C + c) * EN + n] - src[((size_t)b * C + c) * EN + j];
}

// valg[b,d,n,k] = vald[b,d,idx[b,n,k]] + pe[b,d,n,k]
__global__ void gather_addpe_kernel(float* __restrict__ out, const float* __restrict__ src,
                                    const int* __restrict__ idx, const float* __restrict__ pe,
                                    size_t total)
{
    size_t t = (size_t)blockIdx.x * blockDim.x + threadIdx.x;
    if (t >= total) return;
    int k = (int)(t % EK); size_t r = t / EK;
    int n = (int)(r % EN); r /= EN;
    int d = (int)(r % EDIM); int b = (int)(r / EDIM);
    int j = idx[((size_t)b * EN + n) * EK + k];
    out[t] = src[((size_t)b * EDIM + d) * EN + j] + pe[t];
}

// (B,DIM,N,K) -> (B,N,DIM,K) with add: out = transpose(qk + pe)
__global__ void transpose_add_kernel(float* __restrict__ out, const float* __restrict__ qk,
                                     const float* __restrict__ pe, size_t total)
{
    size_t t = (size_t)blockIdx.x * blockDim.x + threadIdx.x;
    if (t >= total) return;
    int k = (int)(t % EK); size_t r = t / EK;
    int d = (int)(r % EDIM); r /= EDIM;
    int n = (int)(r % EN); int b = (int)(r / EN);
    size_t s = (((size_t)b * EDIM + d) * EN + n) * EK + k;
    out[t] = qk[s] + pe[s];
}

// out = a + b2 + c2 elementwise (add_1 + pe, with y_N reshape == flat identity)
__global__ void add3_kernel(float* __restrict__ out, const float* __restrict__ a,
                            const float* __restrict__ b2, const float* __restrict__ c2,
                            size_t total)
{
    size_t t = (size_t)blockIdx.x * blockDim.x + threadIdx.x;
    if (t >= total) return;
    out[t] = a[t] + b2[t] + c2[t];
}

// per (b,n,d): aggN[b,d,n] = sum_k softmax_k(logits[b,n,d,:]) * (vald[b,d,n]+pe[b,d,n,k])
__global__ void softmax_agg_kernel(float* __restrict__ aggN, const float* __restrict__ attN,
                                   const float* __restrict__ vald, const float* __restrict__ pe,
                                   size_t total)
{
    size_t t = (size_t)blockIdx.x * blockDim.x + threadIdx.x;
    if (t >= total) return;
    int d = (int)(t % EDIM); size_t r = t / EDIM;
    int n = (int)(r % EN); int b = (int)(r / EN);
    const float* lg = attN + ((size_t)b * EN + n) * (EDIM * EK) + (size_t)d * EK;
    float mx = lg[0];
#pragma unroll
    for (int k = 1; k < EK; ++k) mx = fmaxf(mx, lg[k]);
    float e[EK]; float s = 0.0f;
#pragma unroll
    for (int k = 0; k < EK; ++k) { e[k] = expf(lg[k] - mx); s += e[k]; }
    float vd = vald[((size_t)b * EDIM + d) * EN + n];
    const float* pp = pe + (((size_t)b * EDIM + d) * EN + n) * EK;
    float accv = 0.0f;
#pragma unroll
    for (int k = 0; k < EK; ++k) accv += (e[k] / s) * (vd + pp[k]);
    aggN[((size_t)b * EDIM + d) * EN + n] = accv;
}

// Wt[r,o,c] = attn_ct_w[c,o,r]   (ctw is (256,64,2))
__global__ void ct_transpose_kernel(float* __restrict__ wt, const float* __restrict__ ctw, size_t total)
{
    size_t t = (size_t)blockIdx.x * blockDim.x + threadIdx.x;
    if (t >= total) return;
    int c = (int)(t % 256); size_t r = t / 256;
    int o = (int)(r % EDIM); int rr = (int)(r / EDIM);
    wt[t] = ctw[((size_t)c * EDIM + o) * 2 + rr];
}

// agg[b,d,2n+r] = sum_k att_r[b,d,n,k] * valg[b,d,n,k]
__global__ void agg_up_kernel(float* __restrict__ agg, const float* __restrict__ att0,
                              const float* __restrict__ att1, const float* __restrict__ vg,
                              size_t total)
{
    size_t t = (size_t)blockIdx.x * blockDim.x + threadIdx.x;
    if (t >= total) return;
    int n = (int)(t % EN); size_t r = t / EN;
    int d = (int)(r % EDIM); int b = (int)(r / EDIM);
    size_t base = (((size_t)b * EDIM + d) * EN + n) * EK;
    float s0 = 0.0f, s1 = 0.0f;
#pragma unroll
    for (int k = 0; k < EK; ++k) {
        float v = vg[base + k];
        s0 += att0[base + k] * v;
        s1 += att1[base + k] * v;
    }
    size_t ob = ((size_t)b * EDIM + d) * ENUP + 2 * (size_t)n;
    agg[ob] = s0; agg[ob + 1] = s1;
}

// x1pre[b,c,m] = y[b,c,m] + identity[b,c,m/2]
__global__ void residual_up_kernel(float* __restrict__ out, const float* __restrict__ y,
                                   const float* __restrict__ iden, size_t total)
{
    size_t t = (size_t)blockIdx.x * blockDim.x + threadIdx.x;
    if (t >= total) return;
    int m = (int)(t % ENUP); size_t r = t / ENUP;
    int c = (int)(r % ECIN); int b = (int)(r / ECIN);
    out[t] = y[t] + iden[((size_t)b * ECIN + c) * EN + (m >> 1)];
}

// cat([x, broadcast(feat)], axis=1): out (B, 1152, NUP)
__global__ void concat_feat_kernel(float* __restrict__ out, const float* __restrict__ x,
                                   const float* __restrict__ feat, size_t total)
{
    size_t t = (size_t)blockIdx.x * blockDim.x + threadIdx.x;
    if (t >= total) return;
    int m = (int)(t % ENUP); size_t r = t / ENUP;
    int c = (int)(r % (ECIN + EFEAT)); int b = (int)(r / (ECIN + EFEAT));
    out[t] = (c < ECIN) ? x[((size_t)b * ECIN + c) * ENUP + m]
                        : feat[(size_t)b * EFEAT + (c - ECIN)];
}

// ---------------------------------------------------------------------------
// Host-side orchestration
// ---------------------------------------------------------------------------
static inline void launch_gemm(const float* W, const float* X, const float* bias, float* out,
                               int O, int C, int S, int relu, int accum, hipStream_t st)
{
    // O % 64 == 0, C % 32 == 0, S % 64 == 0 for every call routed here
    dim3 grid(S / 64, O / 64, EB);
    gemm_wmma_fast<<<grid, dim3(128, 1, 1), 0, st>>>(W, X, bias, out, O, C, S, relu, accum);
}

#define EWG(total) dim3((unsigned)(((total) + 255) / 256)), dim3(256), 0, stream

extern "C" void kernel_launch(void* const* d_in, const int* in_sizes, int n_in,
                              void* d_out, int out_size, void* d_ws, size_t ws_size,
                              hipStream_t stream)
{
    (void)in_sizes; (void)n_in; (void)out_size; (void)ws_size;

    // --- inputs, setup_inputs() insertion order, depth-first through dicts ---
    const float* feat  = (const float*)d_in[0];
    const float* xyz   = (const float*)d_in[1];
    const float* pfeat = (const float*)d_in[2];
    int ii = 3;
#define NXT() ((const float*)d_in[ii++])
    const float* mv_c1w = NXT(); const float* mv_c1b = NXT();
    const float* mv_c2w = NXT(); const float* mv_c2b = NXT();
    const float* mv_scw = NXT(); const float* mv_scb = NXT();
    const float* kw  = NXT(); const float* kb  = NXT();
    const float* qw  = NXT(); const float* qb  = NXT();
    const float* vw  = NXT(); const float* vbi = NXT();
    const float* pc1w = NXT(); const float* pc1b = NXT();
    const float* pbn_g = NXT(); const float* pbn_b = NXT();
    const float* pbn_m = NXT(); const float* pbn_v = NXT();
    const float* pc2w = NXT(); const float* pc2b = NXT();
    const float* an1w = NXT(); const float* an1b = NXT();
    const float* anbn_g = NXT(); const float* anbn_b = NXT();
    const float* anbn_m = NXT(); const float* anbn_v = NXT();
    const float* an2w = NXT(); const float* an2b = NXT();
    const float* ce1w = NXT(); const float* ce1b = NXT();
    const float* ac1w = NXT(); const float* ac1b = NXT();
    const float* abn_g = NXT(); const float* abn_b = NXT();
    const float* abn_m = NXT(); const float* abn_v = NXT();
    const float* ctw = NXT(); const float* ctb = NXT();
    const float* cew = NXT(); const float* ceb = NXT();
    const float* m1_c1w = NXT(); const float* m1_c1b = NXT();
    const float* m1_c2w = NXT(); const float* m1_c2b = NXT();
    const float* m1_scw = NXT(); const float* m1_scb = NXT();
    const float* m2_c1w = NXT(); const float* m2_c1b = NXT();
    const float* m2_c2w = NXT(); const float* m2_c2b = NXT();
    const float* m2_scw = NXT(); const float* m2_scb = NXT();
    const float* m3_c1w = NXT(); const float* m3_c1b = NXT();
    const float* m3_c2w = NXT(); const float* m3_c2b = NXT();
    const float* m3_scw = NXT(); const float* m3_scb = NXT();
    const float* m4c1w = NXT(); const float* m4c1b = NXT();
    const float* m4c2w = NXT(); const float* m4c2b = NXT();
#undef NXT

    // --- workspace bump allocator (256B aligned) ---
    char* wsb = (char*)d_ws;
    size_t cur = 0;
    auto alloc = [&](size_t nfloats) -> float* {
        float* p = (float*)(wsb + cur);
        cur += ((nfloats * 4 + 255) / 256) * 256;
        return p;
    };

    const size_t NK   = (size_t)EN * EK;            // 10240
    const size_t DNK  = (size_t)EDIM * NK;          // 655360
    float* catpf  = alloc((size_t)EB * 256 * EN);
    float* vbuf   = alloc((size_t)EB * ECIN * EN);      // mlp_v out == identity
    float* htmp   = alloc((size_t)EB * ECIN * EN);
    float* keyb   = alloc((size_t)EB * EDIM * EN);
    float* qryb   = alloc((size_t)EB * EDIM * EN);
    float* valdb  = alloc((size_t)EB * EDIM * EN);
    int*   idxb   = (int*)alloc((size_t)EB * EN * EK);
    float* posrel = alloc((size_t)EB * 3 * NK);
    float* posh   = alloc((size_t)EB * EDIM * NK);
    float* pe     = alloc((size_t)EB * DNK);
    float* qkrel  = alloc((size_t)EB * DNK);
    float* qkpeN  = alloc((size_t)EB * DNK);            // reused as add1+pe
    float* BIG    = alloc((size_t)EB * 2048 * 1280);    // hN, then h2
    float* attN   = alloc((size_t)EB * EN * EDIM * EK); // reused as att0
    float* aggN   = alloc((size_t)EB * EDIM * EN);
    float* yN     = alloc((size_t)EB * 1280 * EN);      // reused as att1
    float* valg   = alloc((size_t)EB * DNK);
    float* aggb   = alloc((size_t)EB * EDIM * ENUP);
    float* ybuf   = alloc((size_t)EB * ECIN * ENUP);
    float* x1pre  = alloc((size_t)EB * ECIN * ENUP);
    float* cat1   = alloc((size_t)EB * (ECIN + EFEAT) * ENUP);  // reused as cat2
    float* x1b    = alloc((size_t)EB * ECIN * ENUP);
    float* hb     = alloc((size_t)EB * ECIN * ENUP);
    float* x2b    = alloc((size_t)EB * ECIN * ENUP);
    float* h64    = alloc((size_t)EB * EDIM * ENUP);
    float* fpw    = alloc((size_t)EDIM * 3);  float* fpb = alloc(EDIM);
    float* faw    = alloc((size_t)2048 * EN); float* fab = alloc(2048);
    float* fcw    = alloc((size_t)256 * EDIM);float* fcb = alloc(256);
    float* Wt     = alloc((size_t)2 * EDIM * 256);
    float* add1pe = qkpeN;   // dead after aN_c1 GEMM
    float* h2     = BIG;     // hN dead after aN_c2 GEMM
    float* att0   = attN;    // attN dead after softmax_agg
    float* att1   = yN;      // yN dead after add3
    float* cat2   = cat1;    // cat1 dead after mlp_1
    float* comp   = (float*)d_out;                           // (B,3,NUP)
    float* x3     = (float*)d_out + (size_t)EB * 3 * ENUP;   // (B,128,NUP)

    // ===== value = mlp_res(cat[pf,pf], mlp_v); identity = value =====
    concat_dup_kernel<<<EWG((size_t)EB * 256 * EN)>>>(catpf, pfeat, (size_t)EB * 256 * EN);
    launch_gemm(mv_scw, catpf, mv_scb, vbuf, ECIN, 256, EN, 0, 0, stream);
    launch_gemm(mv_c1w, catpf, mv_c1b, htmp, ECIN, 256, EN, 1, 0, stream);
    launch_gemm(mv_c2w, htmp,  mv_c2b, vbuf, ECIN, ECIN, EN, 0, 1, stream);

    // ===== key / query / value projections =====
    launch_gemm(kw, pfeat, kb,  keyb,  EDIM, ECIN, EN, 0, 0, stream);
    launch_gemm(qw, pfeat, qb,  qryb,  EDIM, ECIN, EN, 0, 0, stream);
    launch_gemm(vw, vbuf,  vbi, valdb, EDIM, ECIN, EN, 0, 0, stream);

    // ===== kNN + grouping =====
    knn_topk_kernel<<<EWG((size_t)EB * EN)>>>(xyz, idxb);
    gather_sub_kernel<<<EWG((size_t)EB * DNK)>>>(qkrel, qryb, keyb, idxb, EDIM, (size_t)EB * DNK);
    gather_sub_kernel<<<EWG((size_t)EB * 3 * NK)>>>(posrel, xyz, xyz, idxb, 3, (size_t)EB * 3 * NK);

    // ===== positional encoding: conv(3->64)+BN+ReLU (tiny: plain), conv(64->64): WMMA =====
    fold_bn_kernel<<<EWG((size_t)EDIM * 3)>>>(pc1w, pc1b, pbn_g, pbn_b, pbn_m, pbn_v, fpw, fpb, EDIM, 3);
    gemm_small<<<EWG((size_t)EB * EDIM * NK)>>>(fpw, posrel, fpb, posh, EDIM, 3, (int)NK, 1,
                                                (size_t)EB * EDIM * NK);
    launch_gemm(pc2w, posh, pc2b, pe, EDIM, EDIM, (int)NK, 0, 0, stream);

    // ===== aN branch: channels = N =====
    transpose_add_kernel<<<EWG((size_t)EB * DNK)>>>(qkpeN, qkrel, pe, (size_t)EB * DNK);
    fold_bn_kernel<<<EWG((size_t)2048 * EN)>>>(an1w, an1b, anbn_g, anbn_b, anbn_m, anbn_v, faw, fab, 2048, EN);
    launch_gemm(faw,  qkpeN, fab,  BIG,  2048, EN,   1280, 1, 0, stream);
    launch_gemm(an2w, BIG,   an2b, attN, EN,   2048, 1280, 0, 0, stream);
    softmax_agg_kernel<<<EWG((size_t)EB * EN * EDIM)>>>(aggN, attN, valdb, pe, (size_t)EB * EN * EDIM);
    launch_gemm(ce1w, aggN, ce1b, yN, 1280, EDIM, EN, 0, 0, stream);

    // add_1 + pe  (y_N reshape is a flat-identity reinterpretation)
    add3_kernel<<<EWG((size_t)EB * DNK)>>>(add1pe, qkrel, yN, pe, (size_t)EB * DNK);

    // ===== main attention: conv(64->256)+BN+ReLU, ConvTranspose(UP=2) =====
    fold_bn_kernel<<<EWG((size_t)256 * EDIM)>>>(ac1w, ac1b, abn_g, abn_b, abn_m, abn_v, fcw, fcb, 256, EDIM);
    launch_gemm(fcw, add1pe, fcb, h2, 256, EDIM, (int)NK, 1, 0, stream);
    ct_transpose_kernel<<<EWG((size_t)2 * EDIM * 256)>>>(Wt, ctw, (size_t)2 * EDIM * 256);
    launch_gemm(Wt,              h2, ctb, att0, EDIM, 256, (int)NK, 0, 0, stream);
    launch_gemm(Wt + EDIM * 256, h2, ctb, att1, EDIM, 256, (int)NK, 0, 0, stream);
    gather_addpe_kernel<<<EWG((size_t)EB * DNK)>>>(valg, valdb, idxb, pe, (size_t)EB * DNK);
    agg_up_kernel<<<EWG((size_t)EB * EDIM * EN)>>>(aggb, att0, att1, valg, (size_t)EB * EDIM * EN);
    launch_gemm(cew, aggb, ceb, ybuf, ECIN, EDIM, ENUP, 0, 0, stream);
    residual_up_kernel<<<EWG((size_t)EB * ECIN * ENUP)>>>(x1pre, ybuf, vbuf, (size_t)EB * ECIN * ENUP);

    // ===== seed generator head =====
    const size_t CATSZ = (size_t)EB * (ECIN + EFEAT) * ENUP;
    concat_feat_kernel<<<EWG(CATSZ)>>>(cat1, x1pre, feat, CATSZ);
    launch_gemm(m1_scw, cat1, m1_scb, x1b, ECIN, ECIN + EFEAT, ENUP, 0, 0, stream);
    launch_gemm(m1_c1w, cat1, m1_c1b, hb,  ECIN, ECIN + EFEAT, ENUP, 1, 0, stream);
    launch_gemm(m1_c2w, hb,   m1_c2b, x1b, ECIN, ECIN,         ENUP, 0, 1, stream);

    launch_gemm(m2_scw, x1b, m2_scb, x2b, ECIN, ECIN, ENUP, 0, 0, stream);
    launch_gemm(m2_c1w, x1b, m2_c1b, h64, EDIM, ECIN, ENUP, 1, 0, stream);
    launch_gemm(m2_c2w, h64, m2_c2b, x2b, ECIN, EDIM, ENUP, 0, 1, stream);

    concat_feat_kernel<<<EWG(CATSZ)>>>(cat2, x2b, feat, CATSZ);
    launch_gemm(m3_scw, cat2, m3_scb, x3, ECIN, ECIN + EFEAT, ENUP, 0, 0, stream);
    launch_gemm(m3_c1w, cat2, m3_c1b, hb, ECIN, ECIN + EFEAT, ENUP, 1, 0, stream);
    launch_gemm(m3_c2w, hb,   m3_c2b, x3, ECIN, ECIN,         ENUP, 0, 1, stream);

    launch_gemm(m4c1w, x3,  m4c1b, h64,  EDIM, ECIN, ENUP, 1, 0, stream);
    gemm_small<<<EWG((size_t)EB * 3 * ENUP)>>>(m4c2w, h64, m4c2b, comp, 3, EDIM, ENUP, 0,
                                               (size_t)EB * 3 * ENUP);
}
